// StochasticDurationPredictor_3951369912642
// MI455X (gfx1250) — matrix operand order repaired and checked by reading, not compiled, not run on hardware
//
#include <hip/hip_runtime.h>
#include <math.h>

// ---------------------------------------------------------------------------
// CDNA5 (gfx1250) implementation of the VITS StochasticDurationPredictor fwd.
// Dominant work: ~36 GEMMs of W[192x192] * X[192 x 16384] -> WMMA f16/f32.
// Everything else is elementwise, sized for wave32.
// ---------------------------------------------------------------------------

typedef __attribute__((ext_vector_type(16))) _Float16 v16h;
typedef __attribute__((ext_vector_type(8)))  float    v8f;

#define BB 32
#define TT 512
#define CC 192
#define LOG2PI 1.8378770664093453f

// ---------------------------------------------------------------------------
// WMMA pointwise conv: Y[b,m,t] = (sum_k W[m,k]*X[b,k,t] + bias[m]) * mask?
// K fixed = 192 (6 steps of 32). One wave -> one 16x16 tile. 4 waves / block.
// T is baked in (TT=512) so every A/B/store address is base + compile-time
// immediate: the B stream becomes 96 independent global_load_b32 with
// constant IOFFSETs (clause-able), no 64-bit address chains.
// A fragment layout (16x32 f16): lane l holds row m = m0 + (l&15);
//   element j<8  -> K = k0 + (l>>4)*8 + j
//   element j>=8 -> K = k0 + 16 + (l>>4)*8 + (j-8)
// B fragment layout (32x16 f16): lane l holds col n = n0 + (l&15);
//   element j<8 -> K = k0 + (l>>4)*8 + j; j>=8 -> K = k0 + 16 + (l>>4)*8 + (j-8)
// C/D layout: VGPR r of lane l -> M = m0 + (l>>4)*8 + r, N = n0 + (l&15)
// ---------------------------------------------------------------------------
__global__ __launch_bounds__(128) void gemm192_wmma_k(
    float* __restrict__ Y, const float* __restrict__ X,
    const float* __restrict__ W, const float* __restrict__ bias,
    const float* __restrict__ maskp, int Mout, int Mstride)
{
  const int lane = threadIdx.x & 31;
  const int wv   = threadIdx.x >> 5;
  const int m0   = (blockIdx.x * 4 + wv) * 16;
  const int n0   = blockIdx.y * 16;
  const int b    = blockIdx.z;
  const int mPad = (Mout + 15) & ~15;
  if (m0 >= mPad) return;                      // uniform per wave: EXEC stays full

  const int half = lane >> 4;
  const int mr   = lane & 15;
  const int t    = n0 + mr;
  // Single B-side base pointer; all K offsets are compile-time immediates.
  const float* XbH = X + (size_t)b * CC * TT + t + (size_t)(half * 8) * TT;
  const int   mrow = m0 + mr;
  const int   mcl  = (mrow < Mout) ? mrow : (Mout - 1);  // clamped, branchless
  const float wsc  = (mrow < Mout) ? 1.0f : 0.0f;        // zero out pad rows
  const float* wrow = W + (size_t)mcl * CC + half * 8;

  v8f acc = {};
  #pragma unroll
  for (int k0 = 0; k0 < 192; k0 += 32) {
    if (k0 + 32 < 192)
      __builtin_prefetch((const void*)(XbH + (k0 + 32) * TT), 0, 1);

    // A: four b128 loads (rows contiguous in K; constant offsets from wrow)
    float4 wa = *(const float4*)(wrow + k0);
    float4 wb = *(const float4*)(wrow + k0 + 4);
    float4 wc = *(const float4*)(wrow + k0 + 16);
    float4 wd = *(const float4*)(wrow + k0 + 20);
    const float a0[8] = {wa.x, wa.y, wa.z, wa.w, wb.x, wb.y, wb.z, wb.w};
    const float a1[8] = {wc.x, wc.y, wc.z, wc.w, wd.x, wd.y, wd.z, wd.w};

    v16h af, bf;
    #pragma unroll
    for (int j = 0; j < 8; ++j) {
      af[j]     = (_Float16)(a0[j] * wsc);
      af[j + 8] = (_Float16)(a1[j] * wsc);
    }
    // B: 16 channel-strided b32 loads, all base + constant immediate
    #pragma unroll
    for (int j = 0; j < 16; ++j)
      bf[j] = (_Float16)XbH[(k0 + j + (j >= 8 ? 8 : 0)) * TT];

    acc = __builtin_amdgcn_wmma_f32_16x16x32_f16(false, af, false, bf,
                                                 (short)0, acc, false, false);
  }

  // Bias: 8 clamped reads, hoisted out of the loop.
  float bv[8];
  #pragma unroll
  for (int r = 0; r < 8; ++r) {
    int m  = m0 + half * 8 + r;
    int mb = (m < Mout) ? m : (Mout - 1);
    bv[r] = bias[mb];
  }
  const float mk = maskp ? maskp[(size_t)b * TT + t] : 1.0f;
  // Stores: single base + constant offsets (m < mPad <= Mstride: in-bounds).
  float* Yb = Y + ((size_t)b * Mstride + (size_t)(m0 + half * 8)) * TT + t;
  #pragma unroll
  for (int r = 0; r < 8; ++r)
    Yb[r * TT] = (acc[r] + bv[r]) * mk;
}

// ---------------------------------------------------------------------------
// Depthwise conv, k=3, dilation d, input pre-multiplied by mask, + bias.
// ---------------------------------------------------------------------------
__global__ void dwconv3_k(float* __restrict__ Y, const float* __restrict__ X,
                          const float* __restrict__ mask,
                          const float* __restrict__ w3, const float* __restrict__ bs,
                          int dil, int T, size_t n)
{
  size_t i = (size_t)blockIdx.x * blockDim.x + threadIdx.x;
  if (i >= n) return;
  int t = (int)(i % T);
  size_t bc = i / T;
  int c = (int)(bc % CC);
  int b = (int)(bc / CC);
  const float* xr = X + bc * T;
  const float* mr = mask + (size_t)b * T;
  float acc = bs[c];
  #pragma unroll
  for (int j = 0; j < 3; ++j) {
    int tt = t + (j - 1) * dil;
    if (tt >= 0 && tt < T) acc += w3[c * 3 + j] * xr[tt] * mr[tt];
  }
  Y[i] = acc;
}

// ---------------------------------------------------------------------------
// Channel LayerNorm (axis=1, C=192) + optional exact GELU.
// One wave32 per (b,t) column; 6 channels / lane; __shfl_xor reductions.
// ---------------------------------------------------------------------------
__global__ __launch_bounds__(32) void lngelu_k(
    float* __restrict__ Y, const float* __restrict__ X,
    const float* __restrict__ g, const float* __restrict__ bta,
    int T, int applyGelu)
{
  int col = blockIdx.x;
  int b = col / T, t = col - b * T;
  const float* xp = X + (size_t)b * CC * T + t;
  float v[6]; float s = 0.f;
  #pragma unroll
  for (int i = 0; i < 6; ++i) { v[i] = xp[(size_t)(threadIdx.x + 32 * i) * T]; s += v[i]; }
  #pragma unroll
  for (int o = 16; o > 0; o >>= 1) s += __shfl_xor(s, o, 32);
  float mean = s * (1.f / 192.f);
  float q = 0.f;
  #pragma unroll
  for (int i = 0; i < 6; ++i) { float d = v[i] - mean; q += d * d; }
  #pragma unroll
  for (int o = 16; o > 0; o >>= 1) q += __shfl_xor(q, o, 32);
  float inv = rsqrtf(q * (1.f / 192.f) + 1e-5f);
  float* yp = Y + (size_t)b * CC * T + t;
  #pragma unroll
  for (int i = 0; i < 6; ++i) {
    int c = threadIdx.x + 32 * i;
    float y = (v[i] - mean) * inv * g[c] + bta[c];
    if (applyGelu) y = 0.5f * y * (1.f + erff(y * 0.70710678118f));
    yp[(size_t)c * T] = y;
  }
}

// ---------------------------------------------------------------------------
// Elementwise helpers.
// ---------------------------------------------------------------------------
__global__ void add_inplace_k(float* __restrict__ x, const float* __restrict__ y, size_t n)
{ size_t i = (size_t)blockIdx.x * blockDim.x + threadIdx.x; if (i < n) x[i] += y[i]; }

__global__ void add3_k(float* __restrict__ z, const float* __restrict__ x,
                       const float* __restrict__ y, size_t n)
{ size_t i = (size_t)blockIdx.x * blockDim.x + threadIdx.x; if (i < n) z[i] = x[i] + y[i]; }

__global__ void mulmask_k(float* __restrict__ x, const float* __restrict__ mask,
                          int C, int T, size_t n)
{
  size_t i = (size_t)blockIdx.x * blockDim.x + threadIdx.x; if (i >= n) return;
  int t = (int)(i % T); int b = (int)(i / ((size_t)C * T));
  x[i] *= mask[(size_t)b * T + t];
}

// y[b,c,t] = wv[c] * x0[b*xstride + t] + bv[c]   (1x1 conv from 1 channel)
__global__ void bcast1_k(float* __restrict__ y, const float* __restrict__ x0,
                         const float* __restrict__ wv, const float* __restrict__ bv,
                         int T, int xstride, size_t n)
{
  size_t i = (size_t)blockIdx.x * blockDim.x + threadIdx.x; if (i >= n) return;
  int t = (int)(i % T);
  size_t bc = i / T;
  int c = (int)(bc % CC);
  int b = (int)(bc / CC);
  y[i] = wv[c] * x0[(size_t)b * xstride + t] + bv[c];
}

__global__ void zero_k(float* p, int n)
{ int i = blockIdx.x * blockDim.x + threadIdx.x; if (i < n) p[i] = 0.f; }

// e = e_q * mask; accumulate eterm[b] += -0.5*(LOG2PI + e^2)*mask
__global__ void ecopy_k(float* __restrict__ z, const float* __restrict__ eq,
                        const float* __restrict__ mask, float* __restrict__ eterm,
                        int T, size_t n2)
{
  size_t i = (size_t)blockIdx.x * blockDim.x + threadIdx.x; if (i >= n2) return;
  int b = (int)(i / (2 * T));
  int t = (int)(i % T);
  float m = mask[(size_t)b * T + t];
  float e = eq[i] * m;
  z[i] = e;
  atomicAdd(&eterm[b], -0.5f * (LOG2PI + e * e) * m);
}

// z = (m2[c] + exp(logs[c])*z)*mask ; ld[b] += logs[c]*mask
__global__ void ea_k(float* __restrict__ z, const float* __restrict__ mask,
                     const float* __restrict__ m2, const float* __restrict__ logs2,
                     float* __restrict__ ld, int T, size_t n2)
{
  size_t i = (size_t)blockIdx.x * blockDim.x + threadIdx.x; if (i >= n2) return;
  int b = (int)(i / (2 * T));
  int r = (int)(i % (2 * T));
  int c = r / T;
  int t = r - c * T;
  float mk = mask[(size_t)b * T + t];
  float lg = logs2[c];
  z[i] = (m2[c] + expf(lg) * z[i]) * mk;
  atomicAdd(&ld[b], lg * mk);
}

// ---------------------------------------------------------------------------
// Rational-quadratic spline flow step (+ the z[:, ::-1] channel flip).
// H holds 29 spline channels (padded to 32) already *mask.
// ---------------------------------------------------------------------------
__global__ void rqs_k(float* __restrict__ zout, const float* __restrict__ zin,
                      const float* __restrict__ H, const float* __restrict__ mask,
                      float* __restrict__ ldacc, int T, int n)
{
  int i = blockIdx.x * blockDim.x + threadIdx.x; if (i >= n) return;
  int b = i / T, t = i - b * T;
  const float s    = 0.0721687836487f;   // 1/sqrt(192)
  const float CNST = 0.541324854612f;    // log(exp(1-MIN_D)-1)
  size_t base = (size_t)b * 32 * T + t;

  float uw[10], uh[10], ud[11];
  #pragma unroll
  for (int k = 0; k < 10; ++k) {
    uw[k] = H[base + (size_t)k * T] * s;
    uh[k] = H[base + (size_t)(k + 10) * T] * s;
  }
  ud[0] = CNST; ud[10] = CNST;
  #pragma unroll
  for (int k = 0; k < 9; ++k) ud[k + 1] = H[base + (size_t)(k + 20) * T];

  float x0 = zin[(size_t)b * 2 * T + t];
  float x1 = zin[(size_t)b * 2 * T + T + t];
  float m  = mask[(size_t)b * T + t];
  bool inside = (x1 >= -5.f) && (x1 <= 5.f);
  float x = fminf(fmaxf(x1, -5.f), 5.f);

  float cw[11], chs[11], wid[10], hei[10], der[11];
  { // widths
    float mx = uw[0];
    #pragma unroll
    for (int k = 1; k < 10; ++k) mx = fmaxf(mx, uw[k]);
    float e[10], sm = 0.f;
    #pragma unroll
    for (int k = 0; k < 10; ++k) { e[k] = expf(uw[k] - mx); sm += e[k]; }
    float cum = 0.f; cw[0] = -5.f;
    #pragma unroll
    for (int k = 0; k < 10; ++k) { cum += 0.001f + 0.99f * e[k] / sm; cw[k + 1] = 10.f * cum - 5.f; }
    cw[10] = 5.f;
    #pragma unroll
    for (int k = 0; k < 10; ++k) wid[k] = cw[k + 1] - cw[k];
  }
  { // heights
    float mx = uh[0];
    #pragma unroll
    for (int k = 1; k < 10; ++k) mx = fmaxf(mx, uh[k]);
    float e[10], sm = 0.f;
    #pragma unroll
    for (int k = 0; k < 10; ++k) { e[k] = expf(uh[k] - mx); sm += e[k]; }
    float cum = 0.f; chs[0] = -5.f;
    #pragma unroll
    for (int k = 0; k < 10; ++k) { cum += 0.001f + 0.99f * e[k] / sm; chs[k + 1] = 10.f * cum - 5.f; }
    chs[10] = 5.f;
    #pragma unroll
    for (int k = 0; k < 10; ++k) hei[k] = chs[k + 1] - chs[k];
  }
  #pragma unroll
  for (int k = 0; k < 11; ++k) {
    float u = ud[k];
    float sp = (u > 20.f) ? u : log1pf(expf(u));
    der[k] = 0.001f + sp;
  }

  int idx = -1;
  #pragma unroll
  for (int k = 0; k < 11; ++k) idx += (x >= cw[k]) ? 1 : 0;
  idx = min(max(idx, 0), 9);

  float icw = cw[idx], ibw = wid[idx], ich = chs[idx], ihh = hei[idx];
  float idl = hei[idx] / wid[idx];
  float d0 = der[idx], d1 = der[idx + 1];
  float th = (x - icw) / ibw;
  float t1 = th * (1.f - th);
  float num = ihh * (idl * th * th + d0 * t1);
  float den = idl + (d0 + d1 - 2.f * idl) * t1;
  float outv = ich + num / den;
  float omt = 1.f - th;
  float dnum = idl * idl * (d1 * th * th + 2.f * idl * t1 + d0 * omt * omt);
  float lad = logf(dnum) - 2.f * logf(den);

  float y1 = inside ? outv : x1;
  lad = inside ? lad : 0.f;
  // y = concat[x0, y1] * mask, then flip channels
  zout[(size_t)b * 2 * T + t]     = y1 * m;
  zout[(size_t)b * 2 * T + T + t] = x0 * m;
  atomicAdd(&ldacc[b], lad * m);
}

// ---------------------------------------------------------------------------
// Between flow chains: sigmoid split, z0 = (w - u), z0l = log(max(z0,1e-5)).
// ---------------------------------------------------------------------------
__device__ __forceinline__ float dev_logsigmoid(float v)
{ return (v > 0.f) ? -log1pf(expf(-v)) : v - log1pf(expf(v)); }

__global__ void postsplit_k(float* __restrict__ z, const float* __restrict__ w,
                            const float* __restrict__ mask,
                            float* __restrict__ ldq, float* __restrict__ ldm,
                            int T, int n)
{
  int i = blockIdx.x * blockDim.x + threadIdx.x; if (i >= n) return;
  int b = i / T, t = i - b * T;
  float m  = mask[(size_t)b * T + t];
  float zu = z[(size_t)b * 2 * T + t];
  float u  = (1.f / (1.f + expf(-zu))) * m;
  float z0 = (w[(size_t)b * T + t] - u) * m;
  atomicAdd(&ldq[b], (dev_logsigmoid(zu) + dev_logsigmoid(-zu)) * m);
  float z0l = logf(fmaxf(z0, 1e-5f)) * m;
  atomicAdd(&ldm[b], -z0l);
  z[(size_t)b * 2 * T + t] = z0l;   // channel 1 left unchanged
}

__global__ void final_k(const float* __restrict__ z, const float* __restrict__ mask,
                        float* __restrict__ nll, int T, size_t n2)
{
  size_t i = (size_t)blockIdx.x * blockDim.x + threadIdx.x; if (i >= n2) return;
  int b = (int)(i / (2 * T));
  int t = (int)(i % T);
  float m = mask[(size_t)b * T + t];
  float v = z[i];
  atomicAdd(&nll[b], 0.5f * (LOG2PI + v * v) * m);
}

__global__ void combine_k(float* __restrict__ out, const float* __restrict__ acc, int Bn)
{
  int b = blockIdx.x * blockDim.x + threadIdx.x;
  if (b < Bn) out[b] = acc[96 + b] - acc[32 + b] + acc[64 + b] - acc[b];
}

// ---------------------------------------------------------------------------
// Host orchestration.
// ---------------------------------------------------------------------------
struct DDSP {
  const float* ln1_b[3]; const float* ln1_g[3];
  const float* ln2_b[3]; const float* ln2_g[3];
  const float* pw_b[3];  const float* pw_w[3];
  const float* sep_b[3]; const float* sep_w[3];
};
struct CFP  { DDSP dds; const float* pre_b; const float* pre_w;
              const float* proj_b; const float* proj_w; };
struct FLP  { CFP cf[4]; const float* ea_logs; const float* ea_m; };

extern "C" void kernel_launch(void* const* d_in, const int* in_sizes, int n_in,
                              void* d_out, int out_size, void* d_ws, size_t ws_size,
                              hipStream_t stream)
{
  (void)in_sizes; (void)n_in; (void)out_size; (void)ws_size;
  const int T = TT;
  const float* x    = (const float*)d_in[0];
  const float* mask = (const float*)d_in[1];
  const float* w_in = (const float*)d_in[2];
  const float* e_q  = (const float*)d_in[3];

  // ---- unpack params in jax-pytree (sorted-key) leaf order -----------------
  int pi = 4;
  auto nx = [&]() { return (const float*)d_in[pi++]; };
  auto ldDDS = [&](DDSP& p) {
    for (int i = 0; i < 3; ++i) p.ln1_b[i] = nx();
    for (int i = 0; i < 3; ++i) p.ln1_g[i] = nx();
    for (int i = 0; i < 3; ++i) p.ln2_b[i] = nx();
    for (int i = 0; i < 3; ++i) p.ln2_g[i] = nx();
    for (int i = 0; i < 3; ++i) p.pw_b[i]  = nx();
    for (int i = 0; i < 3; ++i) p.pw_w[i]  = nx();
    for (int i = 0; i < 3; ++i) p.sep_b[i] = nx();
    for (int i = 0; i < 3; ++i) p.sep_w[i] = nx();
  };
  auto ldFlow = [&](FLP& f) {
    for (int i = 0; i < 4; ++i) {
      ldDDS(f.cf[i].dds);
      f.cf[i].pre_b = nx(); f.cf[i].pre_w = nx();
      f.cf[i].proj_b = nx(); f.cf[i].proj_w = nx();
    }
    f.ea_logs = nx(); f.ea_m = nx();
  };
  DDSP convs;       ldDDS(convs);        // 'convs'
  FLP  flows;       ldFlow(flows);       // 'flows'
  DDSP post_convs;  ldDDS(post_convs);   // 'post_convs'
  FLP  post_flows;  ldFlow(post_flows);  // 'post_flows'
  const float* post_pre_b  = nx(); const float* post_pre_w  = nx();
  const float* post_proj_b = nx(); const float* post_proj_w = nx();
  const float* pre_b       = nx(); const float* pre_w       = nx();
  const float* proj_b      = nx(); const float* proj_w      = nx();

  // ---- workspace layout ----------------------------------------------------
  const size_t N192 = (size_t)BB * CC * T;     // 3.1M floats per 192-ch tensor
  float* ws = (float*)d_ws;
  float* buf0  = ws; ws += N192;
  float* buf1  = ws; ws += N192;
  float* buf2  = ws; ws += N192;
  float* bufHX = ws; ws += N192;
  float* bufG  = ws; ws += N192;
  float* bufP  = ws; ws += (size_t)BB * 32 * T;
  float* zA    = ws; ws += (size_t)BB * 2 * T;
  float* zB    = ws; ws += (size_t)BB * 2 * T;
  float* acc   = ws; ws += 128;                // [ldq|ldm|eterm|nll] x 32

  const int TPB = 256;
  const int g192 = (int)((N192 + TPB - 1) / TPB);
  const int gBT  = (BB * T + TPB - 1) / TPB;
  const int gB2T = (BB * 2 * T + TPB - 1) / TPB;

  auto gemm = [&](float* Y, const float* X, const float* W, const float* bias,
                  int Mout, int Mstride, const float* mk) {
    int mt = (Mout + 15) / 16;
    dim3 g((mt + 3) / 4, T / 16, BB);
    gemm192_wmma_k<<<g, dim3(128), 0, stream>>>(Y, X, W, bias, mk, Mout, Mstride);
  };
  auto dds = [&](float* xb, const DDSP& p, const float* g) {
    if (g) add_inplace_k<<<g192, TPB, 0, stream>>>(xb, g, N192);
    int dil = 1;
    for (int i = 0; i < 3; ++i) {
      dwconv3_k<<<g192, TPB, 0, stream>>>(buf1, xb, mask, p.sep_w[i], p.sep_b[i], dil, T, N192);
      lngelu_k<<<BB * T, 32, 0, stream>>>(buf2, buf1, p.ln1_g[i], p.ln1_b[i], T, 1);
      gemm(buf1, buf2, p.pw_w[i], p.pw_b[i], CC, CC, (const float*)0);
      lngelu_k<<<BB * T, 32, 0, stream>>>(buf2, buf1, p.ln2_g[i], p.ln2_b[i], T, 1);
      add_inplace_k<<<g192, TPB, 0, stream>>>(xb, buf2, N192);
      dil *= 3;
    }
    mulmask_k<<<g192, TPB, 0, stream>>>(xb, mask, CC, T, N192);
  };
  auto flowchain = [&](float*& z, float* zalt, const FLP& f, const float* g, float* ldacc) {
    ea_k<<<gB2T, TPB, 0, stream>>>(z, mask, f.ea_m, f.ea_logs, ldacc, T, (size_t)BB * 2 * T);
    float* cur = z; float* oth = zalt;
    for (int i = 0; i < 4; ++i) {
      const CFP& c = f.cf[i];
      bcast1_k<<<g192, TPB, 0, stream>>>(buf0, cur, c.pre_w, c.pre_b, T, 2 * T, N192);
      dds(buf0, c.dds, g);
      gemm(bufP, buf0, c.proj_w, c.proj_b, 29, 32, mask);
      rqs_k<<<gBT, TPB, 0, stream>>>(oth, cur, bufP, mask, ldacc, T, BB * T);
      float* tmp = cur; cur = oth; oth = tmp;
    }
    z = cur;
  };

  // ---- forward graph -------------------------------------------------------
  zero_k<<<1, 128, 0, stream>>>(acc, 128);

  // hx = proj( dds( pre(x) ) ) * mask
  gemm(buf0, x, pre_w, pre_b, CC, CC, (const float*)0);
  dds(buf0, convs, (const float*)0);
  gemm(bufHX, buf0, proj_w, proj_b, CC, CC, mask);

  // h_w = post_proj( dds( post_pre(w) ) ) * mask ; g = hx + h_w
  bcast1_k<<<g192, TPB, 0, stream>>>(buf0, w_in, post_pre_w, post_pre_b, T, T, N192);
  dds(buf0, post_convs, (const float*)0);
  gemm(buf1, buf0, post_proj_w, post_proj_b, CC, CC, mask);
  add3_k<<<g192, TPB, 0, stream>>>(bufG, bufHX, buf1, N192);

  // e = e_q * mask ; eterm[b] += sum(-0.5*(LOG2PI+e^2)*mask)
  ecopy_k<<<gB2T, TPB, 0, stream>>>(zA, e_q, mask, acc + 64, T, (size_t)BB * 2 * T);

  // posterior flow chain (logdet_q -> acc[0..31])
  float* zc = zA;
  flowchain(zc, zB, post_flows, bufG, acc);

  // split: u = sigmoid(z_u), z0 = (w-u), z0l = log(max(z0,1e-5))
  postsplit_k<<<gBT, TPB, 0, stream>>>(zc, w_in, mask, acc, acc + 32, T, BB * T);

  // main flow chain (logdet -> acc[32..63])
  flowchain(zc, (zc == zA) ? zB : zA, flows, bufHX, acc + 32);

  // nll gaussian term, then combine: out = (nll - logdet) + (eterm - logdet_q)
  final_k<<<gB2T, TPB, 0, stream>>>(zc, mask, acc + 96, T, (size_t)BB * 2 * T);
  combine_k<<<1, BB, 0, stream>>>((float*)d_out, acc, BB);
}
